// ResidualVectorQuantizer_35545149342184
// MI455X (gfx1250) — matrix operand order repaired and checked by reading, not compiled, not run on hardware
//
#include <hip/hip_runtime.h>

typedef __attribute__((ext_vector_type(16))) __bf16 v16bf;
typedef __attribute__((ext_vector_type(8)))  float  v8f;

#define BB   16
#define DD   512
#define TT   2048
#define NCB  8
#define KK   1024
#define RSTRIDE 520   // 512 + 8 halves pad -> 1040B row, rotates LDS banks

// ---------------------------------------------------------------------------
// Prep: codebooks fp32 -> bf16 in workspace, plus c2[k] = sum(c*c) in fp32.
// grid = NCB*KK rows, 128 threads, each thread handles 4 contiguous d.
// ---------------------------------------------------------------------------
__global__ __launch_bounds__(128) void rvq_prep_kernel(
    const float* __restrict__ cb, __bf16* __restrict__ cbbf,
    float* __restrict__ c2)
{
    const int row = blockIdx.x;          // 0 .. NCB*KK-1
    const int tid = threadIdx.x;         // 0 .. 127
    const float* src = cb + (size_t)row * DD + tid * 4;
    __bf16* dst = cbbf + (size_t)row * DD + tid * 4;

    float4 v = *(const float4*)src;
    dst[0] = (__bf16)v.x; dst[1] = (__bf16)v.y;
    dst[2] = (__bf16)v.z; dst[3] = (__bf16)v.w;
    float s = v.x * v.x + v.y * v.y + v.z * v.z + v.w * v.w;

    __shared__ float red[128];
    red[tid] = s;
    __syncthreads();
    for (int off = 64; off; off >>= 1) {
        if (tid < off) red[tid] += red[tid + off];
        __syncthreads();
    }
    if (tid == 0) c2[row] = red[0];
}

// ---------------------------------------------------------------------------
// Main RVQ kernel. One WG (256 thr = 8 waves) owns 32 (b,t) points.
//   wave w: handles BOTH 16-point m-tiles, n-column w (128 codes = 8 n-tiles)
//   -> each B fragment loaded once feeds 2 WMMAs (both m-tiles): halves L2
//      codebook traffic vs. one-m-tile-per-wave mapping.
//   thread owns residual regs: point tid&31, dims (tid>>5)*64 .. +64.
// ---------------------------------------------------------------------------
__global__ __launch_bounds__(256) void rvq_main_kernel(
    const float*  __restrict__ x,
    const float*  __restrict__ cbf32,
    const __bf16* __restrict__ cbbf,
    const float*  __restrict__ c2,
    float* __restrict__ out)
{
    __shared__ __bf16 rbf[32 * RSTRIDE];   // bf16 residual mirror
    __shared__ float  pv[8][32];           // per n-column partial min dist
    __shared__ int    pidx[8][32];         // per n-column partial argmin
    __shared__ int    bidx[32];            // final argmin per point

    const int tid    = threadIdx.x;
    const int lane   = tid & 31;
    const int ncol   = tid >> 5;           // wave id = n-column (128 codes)
    const int plocal = tid & 31;           // owned point
    const int dblk   = (tid >> 5) * 64;    // owned d range base
    const int pg     = blockIdx.x * 32 + plocal;
    const int b      = pg >> 11;
    const int t      = pg & (TT - 1);

    const float* xp = x + (size_t)b * DD * TT + t;

    // residual = x  (fp32 in registers), bf16 mirror in LDS
    float r[64];
#pragma unroll
    for (int j = 0; j < 64; ++j) r[j] = xp[(size_t)(dblk + j) * TT];
#pragma unroll
    for (int j = 0; j < 64; ++j)
        rbf[plocal * RSTRIDE + dblk + j] = (__bf16)r[j];
    __syncthreads();

    // per-lane fragment addressing (ISA WMMA layouts, wave32)
    const int arow0 = (lane & 15);                              // m-tile 0 row
    const int abase0 = arow0 * RSTRIDE + ((lane >> 4) << 3);    // A: +8 halves hi half-wave
    const int abase1 = abase0 + 16 * RSTRIDE;                   // m-tile 1 row
    const int bbase = (lane >> 4) << 3;                         // B: +8 halves hi half-wave
    const int bcode = (lane & 15);

    for (int cb = 0; cb < NCB; ++cb) {
        const __bf16* cbrow = cbbf + (size_t)cb * KK * DD;
        const float*  c2row = c2 + cb * KK;

        float bv[2][8];
        int   bi[2][8];
#pragma unroll
        for (int mt = 0; mt < 2; ++mt)
#pragma unroll
            for (int i = 0; i < 8; ++i) { bv[mt][i] = 3.4e38f; bi[mt][i] = 0; }

        for (int g = 0; g < 4; ++g) {          // 4 groups of 2 n-tiles
            v8f acc[2][2];
            const v8f zero = {0.f, 0.f, 0.f, 0.f, 0.f, 0.f, 0.f, 0.f};
#pragma unroll
            for (int mt = 0; mt < 2; ++mt)
#pragma unroll
                for (int j = 0; j < 2; ++j) acc[mt][j] = zero;

#pragma unroll 2
            for (int dc = 0; dc < 16; ++dc) {  // D = 512 = 16 * 32
                union { v16bf v; uint4 q[2]; } A0, A1;
                A0.q[0] = *(const uint4*)(&rbf[abase0 + dc * 32]);
                A0.q[1] = *(const uint4*)(&rbf[abase0 + dc * 32 + 16]);
                A1.q[0] = *(const uint4*)(&rbf[abase1 + dc * 32]);
                A1.q[1] = *(const uint4*)(&rbf[abase1 + dc * 32 + 16]);
#pragma unroll
                for (int j = 0; j < 2; ++j) {
                    const int code = (ncol * 8 + g * 2 + j) * 16 + bcode;
                    union { v16bf v; uint4 q[2]; } Bm;
                    const __bf16* bp = cbrow + (size_t)code * DD + dc * 32 + bbase;
                    Bm.q[0] = *(const uint4*)(bp);
                    Bm.q[1] = *(const uint4*)(bp + 16);
                    acc[0][j] = __builtin_amdgcn_wmma_f32_16x16x32_bf16(
                        false, A0.v, false, Bm.v, (short)0, acc[0][j], false, false);
                    acc[1][j] = __builtin_amdgcn_wmma_f32_16x16x32_bf16(
                        false, A1.v, false, Bm.v, (short)0, acc[1][j], false, false);
                }
            }
            // distances: c2[k] - 2*dot ; per-lane running argmin (both m-tiles)
#pragma unroll
            for (int j = 0; j < 2; ++j) {
                const int code = (ncol * 8 + g * 2 + j) * 16 + bcode;
                const float cc = c2row[code];
#pragma unroll
                for (int mt = 0; mt < 2; ++mt)
#pragma unroll
                    for (int i = 0; i < 8; ++i) {
                        float d = cc - 2.0f * acc[mt][j][i];
                        if (d < bv[mt][i]) { bv[mt][i] = d; bi[mt][i] = code; }
                    }
            }
        }

        // reduce over the 16 lanes of each half-wave (N within tile)
#pragma unroll
        for (int m = 8; m >= 1; m >>= 1) {
#pragma unroll
            for (int mt = 0; mt < 2; ++mt)
#pragma unroll
                for (int i = 0; i < 8; ++i) {
                    float ov = __shfl_xor(bv[mt][i], m, 32);
                    int   oi = __shfl_xor(bi[mt][i], m, 32);
                    if (ov < bv[mt][i] || (ov == bv[mt][i] && oi < bi[mt][i])) {
                        bv[mt][i] = ov; bi[mt][i] = oi;
                    }
                }
        }
        if ((lane & 15) == 0) {
#pragma unroll
            for (int mt = 0; mt < 2; ++mt) {
                const int rowbase = mt * 16 + ((lane >> 4) << 3);
#pragma unroll
                for (int i = 0; i < 8; ++i) {
                    pv[ncol][rowbase + i]   = bv[mt][i];
                    pidx[ncol][rowbase + i] = bi[mt][i];
                }
            }
        }
        __syncthreads();

        // reduce the 8 n-columns, write index output
        if (tid < 32) {
            float fv = pv[0][tid];
            int   fi = pidx[0][tid];
#pragma unroll
            for (int c = 1; c < 8; ++c) {
                float v2 = pv[c][tid];
                int   i2 = pidx[c][tid];
                if (v2 < fv || (v2 == fv && i2 < fi)) { fv = v2; fi = i2; }
            }
            bidx[tid] = fi;
            const int mypg = blockIdx.x * 32 + tid;
            out[(size_t)BB * DD * TT + (size_t)cb * BB * TT + mypg] = (float)fi;
        }
        __syncthreads();

        // residual update (fp32) + refresh bf16 mirror
        const int best = bidx[plocal];
        const float* qrow = cbf32 + ((size_t)cb * KK + best) * DD + dblk;
#pragma unroll
        for (int j = 0; j < 64; ++j) r[j] -= qrow[j];
#pragma unroll
        for (int j = 0; j < 64; ++j)
            rbf[plocal * RSTRIDE + dblk + j] = (__bf16)r[j];
        __syncthreads();
    }

    // quantized_total = x - final residual
#pragma unroll
    for (int j = 0; j < 64; ++j) {
        const size_t off = (size_t)b * DD * TT + (size_t)(dblk + j) * TT + t;
        out[off] = xp[(size_t)(dblk + j) * TT] - r[j];
    }
}

// ---------------------------------------------------------------------------
extern "C" void kernel_launch(void* const* d_in, const int* in_sizes, int n_in,
                              void* d_out, int out_size, void* d_ws, size_t ws_size,
                              hipStream_t stream) {
    const float* x   = (const float*)d_in[0];   // [B, D, T] fp32
    const float* cbs = (const float*)d_in[1];   // [NCB, K, D] fp32
    float* out = (float*)d_out;                 // [B*D*T] fp32 ++ [NCB*B*T] indices

    __bf16* cbbf = (__bf16*)d_ws;                                     // 8.4 MB
    float*  c2   = (float*)((char*)d_ws + (size_t)NCB * KK * DD * 2); // 32 KB

    rvq_prep_kernel<<<NCB * KK, 128, 0, stream>>>(cbs, cbbf, c2);
    rvq_main_kernel<<<(BB * TT) / 32, 256, 0, stream>>>(x, cbs, cbbf, c2, out);
}
